// Glm4vResampler_82755429859782
// MI455X (gfx1250) — matrix operand order repaired and checked by reading, not compiled, not run on hardware
//
#include <hip/hip_runtime.h>

// ---------------- problem constants (from reference module) ----------------
constexpr int E_   = 4096;
constexpr int NH_  = 32;
constexpr int HD_  = 128;          // E / NH
constexpr int B_   = 2;
constexpr int TE_  = 8, HE_ = 32, WE_ = 32;
constexpr int L_   = TE_ * HE_ * WE_;   // 8192
constexpr int VD_  = 1536;
constexpr int NQ_  = 64;

// ---------------- types / helpers ----------------
typedef __attribute__((ext_vector_type(16))) __bf16 v16bf;
typedef __attribute__((ext_vector_type(8)))  float  v8f;
typedef __attribute__((ext_vector_type(4)))  unsigned int v4u;
typedef __attribute__((ext_vector_type(8)))  unsigned int v8u;
typedef __attribute__((address_space(3)))    unsigned short lds_us;

struct Frag2Q { uint4 lo, hi; };

__device__ __forceinline__ v16bf frag_cast(uint4 lo, uint4 hi) {
  Frag2Q f; f.lo = lo; f.hi = hi;
  return __builtin_bit_cast(v16bf, f);
}

__device__ __forceinline__ unsigned short f2bf(float f) {
  unsigned u = __float_as_uint(f);
  u += 0x7FFFu + ((u >> 16) & 1u);       // round-to-nearest-even
  return (unsigned short)(u >> 16);
}

// Issue one TDM 2D tile load: 64 rows x 32 bf16 elems, row stride ldbElems,
// into LDS at byte address ldsAddr. Groups 2/3 omitted (2D tensor).
// D# group0: count=1 | lds_addr | global_addr | type=2
// D# group1: data_size=2B, tensor dims large (no clip), tile 32x64, dim0 stride.
__device__ __forceinline__ void tdm_load_b_tile(unsigned long long gaddr,
                                                unsigned ldsAddr, int ldbElems) {
  v4u g0;
  g0[0] = 1u;                                            // count = 1 (valid)
  g0[1] = ldsAddr;                                       // lds_addr (bytes)
  g0[2] = (unsigned)(gaddr & 0xFFFFFFFFull);             // global_addr[31:0]
  g0[3] = (unsigned)((gaddr >> 32) & 0x01FFFFFFull) | 0x80000000u; // [56:32] | type=2
  v8u g1;
  g1[0] = 0x10000u;                                      // data_size = 1 (2 bytes)
  g1[1] = 0xFFFF0000u;                                   // tensor_dim0[15:0]
  g1[2] = 0xFFFF7FFFu;                                   // tensor_dim0[31:16] | tensor_dim1[15:0]
  g1[3] = 0x00207FFFu;                                   // tensor_dim1[31:16] | tile_dim0=32
  g1[4] = 64u;                                           // tile_dim1 = 64 (tile_dim2 = 0)
  g1[5] = (unsigned)ldbElems;                            // tensor_dim0_stride[31:0]
  g1[6] = 0u;                                            // stride[47:32] | dim1_stride lo
  g1[7] = 0u;
  asm volatile("tensor_load_to_lds %0, %1" :: "s"(g0), "s"(g1) : "memory");
}

// ---------------- generic batched strided bf16 GEMM ----------------
// C[m,n] = alpha * sum_k A[m,k] * Bw[n,k]  (+ bias[n])
// Block: 256 threads (8 waves). Block tile 256(M) x 64(N); each wave 32x64.
// B tiles (64x32) DMA'd to double-buffered LDS by the Tensor Data Mover.
__global__ __launch_bounds__(256)
void k_gemm(const unsigned short* __restrict__ A, long long aO0, long long aO1, int lda,
            const unsigned short* __restrict__ Bw, long long bO0, long long bO1, int ldb,
            void* __restrict__ Cp, long long cO0, long long cO1, int ldc, int cBf16,
            const float* __restrict__ bias, float alpha,
            int M, int N, int K, int bdiv)
{
  __shared__ unsigned short ldsB[2][64 * 32];   // 2 x 4 KB, TDM destination

  const int tid  = threadIdx.x;
  const int lane = tid & 31;
  const int wv   = tid >> 5;

  const int z  = blockIdx.z;
  const int zi = z / bdiv;
  const int zj = z - zi * bdiv;
  const unsigned short* Ab = A  + (long long)zi * aO0 + (long long)zj * aO1;
  const unsigned short* Bb = Bw + (long long)zi * bO0 + (long long)zj * bO1;
  const long long cBase    = (long long)zi * cO0 + (long long)zj * cO1;

  const int n0 = blockIdx.x * 64;
  const int m0 = blockIdx.y * 256 + wv * 32;

  // A layout: lane holds row M = lane&15; K halves [baseA..+8) U [baseA+16..+24).
  int rowA0 = m0 +      (lane & 15); if (rowA0 > M - 1) rowA0 = M - 1;
  int rowA1 = m0 + 16 + (lane & 15); if (rowA1 > M - 1) rowA1 = M - 1;
  const int baseA = (lane >> 4) << 3;              // 0 or 8
  const int baseB = (lane >> 4) << 3;              // 0 or 8 (B: same K halves as A)
  const unsigned short* pA0 = Ab + (size_t)rowA0 * lda + baseA;
  const unsigned short* pA1 = Ab + (size_t)rowA1 * lda + baseA;

  const unsigned long long gB = (unsigned long long)(Bb + (size_t)n0 * ldb);
  const unsigned ldsBase = (unsigned)(unsigned long long)(lds_us*)&ldsB[0][0];

  v8f zero = {};
  v8f acc[2][4];
  #pragma unroll
  for (int i = 0; i < 2; ++i)
    #pragma unroll
    for (int j = 0; j < 4; ++j) acc[i][j] = zero;

  const int nsteps = K >> 5;                       // K / 32
  // prologue: DMA first B tile
  if (wv == 0) {
    tdm_load_b_tile(gB, ldsBase, ldb);
    __builtin_amdgcn_s_wait_tensorcnt(0);
  }
  __syncthreads();

  const int nl = lane & 15;
  for (int s = 0; s < nsteps; ++s) {
    const int cur = s & 1;
    // DMA next tile into the other buffer (its readers finished one barrier ago)
    if (s + 1 < nsteps && wv == 0) {
      tdm_load_b_tile(gB + (unsigned long long)(s + 1) * 64u,
                      ldsBase + (unsigned)((s + 1) & 1) * 4096u, ldb);
    }

    const unsigned short* bufp = &ldsB[cur][0];
    const int kh = s * 32;

    // B fragments: 4 distinct register sets, then a WMMA burst.
    v16bf bfr[4];
    #pragma unroll
    for (int j = 0; j < 4; ++j) {
      const unsigned short* q = bufp + (j * 16 + nl) * 32 + baseB;
      bfr[j] = frag_cast(*(const uint4*)q, *(const uint4*)(q + 16));
    }
    const v16bf a0 = frag_cast(*(const uint4*)(pA0 + kh), *(const uint4*)(pA0 + kh + 16));
    const v16bf a1 = frag_cast(*(const uint4*)(pA1 + kh), *(const uint4*)(pA1 + kh + 16));
    if (s + 1 < nsteps) {                          // uniform branch
      __builtin_prefetch(pA0 + kh + 32, 0, 3);     // global_prefetch_b8
      __builtin_prefetch(pA1 + kh + 32, 0, 3);
    }

    #pragma unroll
    for (int j = 0; j < 4; ++j) {
      acc[0][j] = __builtin_amdgcn_wmma_f32_16x16x32_bf16(false, a0, false, bfr[j],
                                                          (short)0, acc[0][j], false, false);
      acc[1][j] = __builtin_amdgcn_wmma_f32_16x16x32_bf16(false, a1, false, bfr[j],
                                                          (short)0, acc[1][j], false, false);
    }

    if (wv == 0) __builtin_amdgcn_s_wait_tensorcnt(0);  // next buffer resident
    __syncthreads();
  }

  // Epilogue: D layout — lane holds N = lane&15, VGPR r holds M = r + 8*(lane>>4).
  const int col = lane & 15;
  const int hh  = lane >> 4;
  #pragma unroll
  for (int mt = 0; mt < 2; ++mt) {
    #pragma unroll
    for (int j = 0; j < 4; ++j) {
      const int nG = n0 + j * 16 + col;
      const float bval = bias ? bias[nG] : 0.f;
      #pragma unroll
      for (int r = 0; r < 8; ++r) {
        const int m = m0 + mt * 16 + r + hh * 8;
        if (m < M) {
          const long long off = cBase + (long long)m * ldc + nG;
          const float v = acc[mt][j][r] * alpha + bval;
          if (cBf16) ((unsigned short*)Cp)[off] = f2bf(v);
          else       ((float*)Cp)[off] = v;
        }
      }
    }
  }
}

// ---------------- f32 -> bf16 convert ----------------
__global__ __launch_bounds__(256)
void k_cvt(const float* __restrict__ in, unsigned short* __restrict__ out, size_t n) {
  size_t i = (size_t)blockIdx.x * 256 + threadIdx.x;
  if (i < n) out[i] = f2bf(in[i]);
}

// transpose-convert: out[f*E + e] = in[e*E + f]   (final proj is not NT)
__global__ __launch_bounds__(256)
void k_cvt_T(const float* __restrict__ in, unsigned short* __restrict__ out) {
  size_t i = (size_t)blockIdx.x * 256 + threadIdx.x;
  size_t n = (size_t)E_ * E_;
  if (i < n) {
    int f = (int)(i / E_);
    int e = (int)(i - (size_t)f * E_);
    out[i] = f2bf(in[(size_t)e * E_ + f]);
  }
}

// ---------------- fused LayerNorm + transpose + RoPE ----------------
__global__ __launch_bounds__(256)
void k_ln_rope(const float* __restrict__ kvraw, const float* __restrict__ g,
               const float* __restrict__ bta, unsigned short* __restrict__ kv_out,
               unsigned short* __restrict__ rot_out)
{
  __shared__ float sh[256];
  const int row = blockIdx.x;
  const int b = row / L_;
  const int l = row - b * L_;
  const float* x = kvraw + (size_t)row * E_;
  const int tid = threadIdx.x;

  float s = 0.f;
  for (int i = tid; i < E_; i += 256) s += x[i];
  sh[tid] = s; __syncthreads();
  for (int st = 128; st; st >>= 1) { if (tid < st) sh[tid] += sh[tid + st]; __syncthreads(); }
  const float mean = sh[0] * (1.f / E_); __syncthreads();

  float s2 = 0.f;
  for (int i = tid; i < E_; i += 256) { float d = x[i] - mean; s2 += d * d; }
  sh[tid] = s2; __syncthreads();
  for (int st = 128; st; st >>= 1) { if (tid < st) sh[tid] += sh[tid + st]; __syncthreads(); }
  const float inv = rsqrtf(sh[0] * (1.f / E_) + 1e-6f);

  const int t  = l / (HE_ * WE_);
  const int hp = (l / WE_) % HE_;
  const int wp = l % WE_;
  const float fs = 10.f / 3.f;                    // REF_FPS / FPS
  const float pt = (float)t * fs;
  const float nmax = (float)((int)((TE_ - 1) * fs));   // max_t - 1 = 23
  float lo = floorf(pt), hi = ceilf(pt);
  lo = fminf(fmaxf(lo, 0.f), nmax);
  hi = fminf(fmaxf(hi, 0.f), nmax);
  const float wgt  = pt - lo;
  const float teff = lo * (1.f - wgt) + hi * wgt;

  const size_t ob = ((size_t)l * B_ + b) * E_;
  const float LOG1E4 = 9.210340371976184f;        // ln(10000)
  const int f0 = tid * 16;                        // E/256 = 16 elems, 8 pairs
  #pragma unroll
  for (int p = 0; p < 8; ++p) {
    const int f = f0 + 2 * p;
    const float y1 = (x[f]     - mean) * inv * g[f]     + bta[f];
    const float y2 = (x[f + 1] - mean) * inv * g[f + 1] + bta[f + 1];
    kv_out[ob + f]     = f2bf(y1);
    kv_out[ob + f + 1] = f2bf(y2);
    const int c = f;   // even column of concatenated [t(1024)|h(1536)|w(1536)] freqs
    float ang;
    if (c < 1024)      ang = teff      * expf(-((float)(c & 511)          * (2.f / 1024.f)) * LOG1E4);
    else if (c < 2560) ang = (float)hp * expf(-((float)((c - 1024) % 768) * (2.f / 1536.f)) * LOG1E4);
    else               ang = (float)wp * expf(-((float)((c - 2560) % 768) * (2.f / 1536.f)) * LOG1E4);
    const float cs = cosf(ang), sn = sinf(ang);
    rot_out[ob + f]     = f2bf(y1 * cs - y2 * sn);
    rot_out[ob + f + 1] = f2bf(y1 * sn + y2 * cs);
  }
}

// ---------------- plain row LayerNorm (optional (q,b)->(b,q) transpose) ----------------
__global__ __launch_bounds__(256)
void k_ln_rows(const float* __restrict__ in, const float* __restrict__ g,
               const float* __restrict__ bta, unsigned short* __restrict__ out,
               int d0, int d1, int doT)
{
  __shared__ float sh[256];
  const int row = blockIdx.x;
  const float* x = in + (size_t)row * E_;
  int orow = row;
  if (doT) { int i0 = row / d1, i1 = row - i0 * d1; orow = i1 * d0 + i0; }
  unsigned short* o = out + (size_t)orow * E_;
  const int tid = threadIdx.x;

  float s = 0.f;
  for (int i = tid; i < E_; i += 256) s += x[i];
  sh[tid] = s; __syncthreads();
  for (int st = 128; st; st >>= 1) { if (tid < st) sh[tid] += sh[tid + st]; __syncthreads(); }
  const float mean = sh[0] * (1.f / E_); __syncthreads();

  float s2 = 0.f;
  for (int i = tid; i < E_; i += 256) { float d = x[i] - mean; s2 += d * d; }
  sh[tid] = s2; __syncthreads();
  for (int st = 128; st; st >>= 1) { if (tid < st) sh[tid] += sh[tid + st]; __syncthreads(); }
  const float inv = rsqrtf(sh[0] * (1.f / E_) + 1e-6f);

  for (int i = tid; i < E_; i += 256)
    o[i] = f2bf((x[i] - mean) * inv * g[i] + bta[i]);
}

// ---------------- softmax over L (row = (b,h,q)) ----------------
__global__ __launch_bounds__(256)
void k_softmax(const float* __restrict__ S, unsigned short* __restrict__ P) {
  __shared__ float sh[256];
  const size_t row = blockIdx.x;
  const float* x = S + row * (size_t)L_;
  unsigned short* o = P + row * (size_t)L_;
  const int tid = threadIdx.x;

  float mx = -3.0e38f;
  for (int i = tid; i < L_; i += 256) mx = fmaxf(mx, x[i]);
  sh[tid] = mx; __syncthreads();
  for (int st = 128; st; st >>= 1) { if (tid < st) sh[tid] = fmaxf(sh[tid], sh[tid + st]); __syncthreads(); }
  mx = sh[0]; __syncthreads();

  float sum = 0.f;
  for (int i = tid; i < L_; i += 256) sum += expf(x[i] - mx);
  sh[tid] = sum; __syncthreads();
  for (int st = 128; st; st >>= 1) { if (tid < st) sh[tid] += sh[tid + st]; __syncthreads(); }
  const float invs = 1.f / sh[0];

  for (int i = tid; i < L_; i += 256) o[i] = f2bf(expf(x[i] - mx) * invs);
}

// ---------------- vp (L,B,E) -> vpT (B,NH,HD,L) ----------------
__global__ __launch_bounds__(256)
void k_vpT(const unsigned short* __restrict__ vp, unsigned short* __restrict__ vpT) {
  size_t i = (size_t)blockIdx.x * 256 + threadIdx.x;   // output index (b,h,d,l)
  const size_t total = (size_t)B_ * NH_ * HD_ * L_;
  if (i >= total) return;
  const int l = (int)(i % L_);
  size_t r = i / L_;
  const int d = (int)(r % HD_); r /= HD_;
  const int h = (int)(r % NH_);
  const int b = (int)(r / NH_);
  vpT[i] = vp[((size_t)l * B_ + b) * E_ + (size_t)h * HD_ + d];
}

// ---------------- workspace layout (bytes) ----------------
constexpr size_t SZ_CUBE  = (size_t)B_ * L_ * VD_;        // u16 elems
constexpr size_t SZ_WKV   = (size_t)E_ * VD_;
constexpr size_t SZ_WIN   = (size_t)3 * E_ * E_;
constexpr size_t SZ_WMAT  = (size_t)E_ * E_;
constexpr size_t SZ_KVRAW = (size_t)B_ * L_ * E_;         // f32 elems
constexpr size_t SZ_KV    = (size_t)L_ * B_ * E_;         // u16 elems
constexpr size_t SZ_S     = (size_t)B_ * NH_ * NQ_ * L_;  // f32 elems
constexpr size_t SZ_SMALL = (size_t)NQ_ * E_;             // u16 elems

constexpr size_t O_CUBE   = 0;
constexpr size_t O_WKV    = O_CUBE   + 2 * SZ_CUBE;
constexpr size_t O_WIN    = O_WKV    + 2 * SZ_WKV;
constexpr size_t O_WO     = O_WIN    + 2 * SZ_WIN;
constexpr size_t O_WPT    = O_WO     + 2 * SZ_WMAT;
constexpr size_t O_KVRAW  = O_WPT    + 2 * SZ_WMAT;       // f32; recycled for S/P
constexpr size_t O_S      = O_KVRAW;                      // reuse: 134 MB f32
constexpr size_t O_P      = O_S      + 4 * SZ_S;          // reuse: 67 MB bf16
constexpr size_t O_KV     = O_KVRAW  + 4 * SZ_KVRAW;
constexpr size_t O_KVROT  = O_KV     + 2 * SZ_KV;
constexpr size_t O_KP     = O_KVROT  + 2 * SZ_KV;
constexpr size_t O_VP     = O_KP     + 2 * SZ_KV;
constexpr size_t O_VPT    = O_VP     + 2 * SZ_KV;
constexpr size_t O_Q      = O_VPT    + 2 * SZ_KV;
constexpr size_t O_QP     = O_Q      + 2 * SZ_SMALL;
constexpr size_t O_CTX    = O_QP     + 2 * SZ_SMALL;
constexpr size_t O_OUT1   = O_CTX    + 2 * (size_t)NQ_ * B_ * E_;
constexpr size_t O_OUT1LN = O_OUT1   + 4 * (size_t)NQ_ * B_ * E_;

extern "C" void kernel_launch(void* const* d_in, const int* in_sizes, int n_in,
                              void* d_out, int out_size, void* d_ws, size_t ws_size,
                              hipStream_t stream) {
  const float* cube     = (const float*)d_in[0];
  const float* query    = (const float*)d_in[1];
  const float* wkv_f    = (const float*)d_in[2];
  const float* ln_q_w   = (const float*)d_in[3];
  const float* ln_q_b   = (const float*)d_in[4];
  const float* ln_kv_w  = (const float*)d_in[5];
  const float* ln_kv_b  = (const float*)d_in[6];
  const float* ln_po_w  = (const float*)d_in[7];
  const float* ln_po_b  = (const float*)d_in[8];
  const float* win_f    = (const float*)d_in[9];
  const float* b_in     = (const float*)d_in[10];
  const float* wo_f     = (const float*)d_in[11];
  const float* bo       = (const float*)d_in[12];
  const float* proj_f   = (const float*)d_in[13];
  float* out            = (float*)d_out;

  char* ws = (char*)d_ws;
  auto U16 = [&](size_t off) { return (unsigned short*)(ws + off); };
  auto F32 = [&](size_t off) { return (float*)(ws + off); };

  const float inv_sqrt_hd = 0.08838834764831845f;   // 1/sqrt(128)

  // 1) convert weights + activations to bf16
  k_cvt<<<(unsigned)((SZ_CUBE + 255) / 256), 256, 0, stream>>>(cube, U16(O_CUBE), SZ_CUBE);
  k_cvt<<<(unsigned)((SZ_WKV  + 255) / 256), 256, 0, stream>>>(wkv_f, U16(O_WKV), SZ_WKV);
  k_cvt<<<(unsigned)((SZ_WIN  + 255) / 256), 256, 0, stream>>>(win_f, U16(O_WIN), SZ_WIN);
  k_cvt<<<(unsigned)((SZ_WMAT + 255) / 256), 256, 0, stream>>>(wo_f, U16(O_WO), SZ_WMAT);
  k_cvt_T<<<(unsigned)((SZ_WMAT + 255) / 256), 256, 0, stream>>>(proj_f, U16(O_WPT));

  // 2) kv = cube @ Wkv^T  -> f32 (B*L, E)
  k_gemm<<<dim3(E_ / 64, (B_ * L_) / 256, 1), 256, 0, stream>>>(
      U16(O_CUBE), 0, 0, VD_, U16(O_WKV), 0, 0, VD_,
      (void*)F32(O_KVRAW), 0, 0, E_, 0, nullptr, 1.f, B_ * L_, E_, VD_, 1);

  // 3) fused LN + transpose(L,B,E) + RoPE -> kv_bf16, kvrot_bf16
  k_ln_rope<<<B_ * L_, 256, 0, stream>>>(F32(O_KVRAW), ln_kv_w, ln_kv_b,
                                         U16(O_KV), U16(O_KVROT));

  // 4) q = LN(query) -> bf16 (64, E)
  k_ln_rows<<<NQ_, 256, 0, stream>>>(query, ln_q_w, ln_q_b, U16(O_Q), NQ_, 1, 0);

  // 5) qp = q @ Wq^T + bq     (identical across b; compute once)
  k_gemm<<<dim3(E_ / 64, 1, 1), 256, 0, stream>>>(
      U16(O_Q), 0, 0, E_, U16(O_WIN), 0, 0, E_,
      (void*)U16(O_QP), 0, 0, E_, 1, b_in, 1.f, NQ_, E_, E_, 1);

  // 6) kp = kv_rot @ Wk^T + bk   -> bf16 (L,B,E)
  k_gemm<<<dim3(E_ / 64, (L_ * B_) / 256, 1), 256, 0, stream>>>(
      U16(O_KVROT), 0, 0, E_, U16(O_WIN) + (size_t)E_ * E_, 0, 0, E_,
      (void*)U16(O_KP), 0, 0, E_, 1, b_in + E_, 1.f, L_ * B_, E_, E_, 1);

  // 7) vp = kv @ Wv^T + bv   -> bf16 (L,B,E), then transpose to (B,NH,HD,L)
  k_gemm<<<dim3(E_ / 64, (L_ * B_) / 256, 1), 256, 0, stream>>>(
      U16(O_KV), 0, 0, E_, U16(O_WIN) + (size_t)2 * E_ * E_, 0, 0, E_,
      (void*)U16(O_VP), 0, 0, E_, 1, b_in + 2 * E_, 1.f, L_ * B_, E_, E_, 1);
  {
    size_t total = (size_t)B_ * NH_ * HD_ * L_;
    k_vpT<<<(unsigned)((total + 255) / 256), 256, 0, stream>>>(U16(O_VP), U16(O_VPT));
  }

  // 8) scores[b,h,q,l] = (qp[q,h,:] . kp[l,b,h,:]) / sqrt(HD)  -> f32
  k_gemm<<<dim3(L_ / 64, 1, B_ * NH_), 256, 0, stream>>>(
      U16(O_QP), 0, (long long)HD_, E_,
      U16(O_KP), (long long)E_, (long long)HD_, B_ * E_,
      (void*)F32(O_S), (long long)NH_ * NQ_ * L_, (long long)NQ_ * L_, L_, 0,
      nullptr, inv_sqrt_hd, NQ_, L_, HD_, NH_);

  // 9) softmax over l -> P bf16
  k_softmax<<<B_ * NH_ * NQ_, 256, 0, stream>>>(F32(O_S), U16(O_P));

  // 10) ctx[q,b,h*128+d] = sum_l P[b,h,q,l] * vpT[b,h,d,l]  -> bf16 (Nq,B,E)
  k_gemm<<<dim3(HD_ / 64, 1, B_ * NH_), 256, 0, stream>>>(
      U16(O_P),  (long long)NH_ * NQ_ * L_, (long long)NQ_ * L_, L_,
      U16(O_VPT), (long long)NH_ * HD_ * L_, (long long)HD_ * L_, L_,
      (void*)U16(O_CTX), (long long)E_, (long long)HD_, B_ * E_, 1,
      nullptr, 1.f, NQ_, HD_, L_, NH_);

  // 11) out1 = ctx @ Wo^T + bo  -> f32 rows (q,b)
  k_gemm<<<dim3(E_ / 64, 1, 1), 256, 0, stream>>>(
      U16(O_CTX), 0, 0, E_, U16(O_WO), 0, 0, E_,
      (void*)F32(O_OUT1), 0, 0, E_, 0, bo, 1.f, NQ_ * B_, E_, E_, 1);

  // 12) transpose to (b,q) + LN_post -> bf16
  k_ln_rows<<<NQ_ * B_, 256, 0, stream>>>(F32(O_OUT1), ln_po_w, ln_po_b,
                                          U16(O_OUT1LN), NQ_, B_, 1);

  // 13) final: out = out1ln @ proj   (proj pre-transposed) -> f32 (B,Nq,E)
  k_gemm<<<dim3(E_ / 64, 1, 1), 256, 0, stream>>>(
      U16(O_OUT1LN), 0, 0, E_, U16(O_WPT), 0, 0, E_,
      (void*)out, 0, 0, E_, 0, nullptr, 1.f, B_ * NQ_, E_, E_, 1);
}
